// HyenaOperator1D_66279935312021
// MI455X (gfx1250) — matrix operand order repaired and checked by reading, not compiled, not run on hardware
//
#include <hip/hip_runtime.h>

// ---------------------------------------------------------------------------
// Types for CDNA5 WMMA (wave32): v_wmma_f32_16x16x32_bf16
// ---------------------------------------------------------------------------
typedef __attribute__((ext_vector_type(16))) __bf16 v16bf;
typedef __attribute__((ext_vector_type(8)))  __bf16 v8bf;
typedef __attribute__((ext_vector_type(8)))  float  v8f;

__device__ __forceinline__ v16bf cat8(v8bf a, v8bf b) {
  return __builtin_shufflevector(a, b, 0,1,2,3,4,5,6,7,8,9,10,11,12,13,14,15);
}
__device__ __forceinline__ v8f wmma_bf16(v16bf a, v16bf b, v8f c) {
  // 8 args: (neg_a, A, neg_b, B, c_mod, C, reuse_a, reuse_b)
  return __builtin_amdgcn_wmma_f32_16x16x32_bf16(false, a, false, b, (short)0, c,
                                                 false, false);
}

// Problem constants
#define Bz 4
#define Lz 4096
#define Dz 1024
#define WIDTHz 3072
#define Mz (Bz * Lz)        // 16384
#define KPz 4128            // padded filter length: 16 zeros + 4096 + 16 zeros
#define VPz 4352            // padded v length per channel: 256 zeros + 4096

// ---------------------------------------------------------------------------
// K0: transpose + fp32->bf16 convert weights: src[K][N] -> dst[N][K]
// ---------------------------------------------------------------------------
__global__ __launch_bounds__(256) void transpose_cvt(const float* __restrict__ src,
                                                     __bf16* __restrict__ dst,
                                                     int K, int N) {
  int idx = blockIdx.x * 256 + threadIdx.x;
  if (idx >= K * N) return;
  int k = idx / N, n = idx % N;
  dst[(size_t)n * K + k] = (__bf16)src[idx];
}

// ---------------------------------------------------------------------------
// K1: implicit filter MLP -> krev (bf16, per-channel, front/back zero-padded)
// Reference conv taps: out[t] = sum_tau v[t-tau] * k[L-1-tau]  (torch xcorr,
// left pad L-1). With kp[m] = k_lag[m-16], k_lag[tau] = k[L-1-tau], and
// krev[n] = kp[KP-1-n]  ==>  krev[n] = k[n-16]  for n in [16, 16+L).
// ---------------------------------------------------------------------------
__global__ __launch_bounds__(256) void gen_filter(
    const float* __restrict__ W1, const float* __restrict__ b1,
    const float* __restrict__ f1, const float* __restrict__ W2,
    const float* __restrict__ b2, const float* __restrict__ f2,
    const float* __restrict__ W3, const float* __restrict__ b3,
    __bf16* __restrict__ krev) {
  __shared__ float pe[3];
  __shared__ float h1s[64];
  __shared__ float h2s[64];
  const int t = blockIdx.x;
  const int tid = threadIdx.x;
  if (tid == 0) {
    float tn = (float)t / 4095.0f;
    float ang = 2.0f * 3.14159265358979323846f * (float)t / 4096.0f;
    float ph = 1e-4f * ang;
    pe[0] = tn;
    pe[1] = cosf(ph);
    pe[2] = -sinf(ph);
  }
  __syncthreads();
  if (tid < 64) {
    float s = pe[0] * W1[tid] + pe[1] * W1[64 + tid] + pe[2] * W1[128 + tid] + b1[tid];
    h1s[tid] = sinf(s * f1[tid]);
  }
  __syncthreads();
  if (tid < 64) {
    float s = b2[tid];
#pragma unroll 8
    for (int i = 0; i < 64; ++i) s += h1s[i] * W2[i * 64 + tid];
    h2s[tid] = sinf(s * f2[tid]);
  }
  __syncthreads();
  const float tn = (float)t / 4095.0f;
  const float dmax = -15.350567286626975f;  // log(0.01)/0.3
  const float dmin = -3.0701134573253950f;  // log(0.01)/1.5
  for (int d = tid; d < Dz; d += 256) {
    float s = b3[d];
#pragma unroll 8
    for (int i = 0; i < 64; ++i) s += h2s[i] * W3[i * Dz + d];
    float delta = fabsf(dmax + (dmin - dmax) * ((float)d / 1023.0f));
    float kv = s * expf(-tn * delta);
    krev[(size_t)d * KPz + 16 + t] = (__bf16)kv;
  }
  if (blockIdx.x == 0) {  // zero the 16-entry pads at both ends of every row
    for (int j = tid; j < Dz * 32; j += 256) {
      int d = j >> 5, e = j & 31;
      int n = (e < 16) ? e : (4096 + e);  // [0,16) and [4112,4128)
      krev[(size_t)d * KPz + n] = (__bf16)0.0f;
    }
  }
}

// ---------------------------------------------------------------------------
// K2: u = x @ W_in + b_in  (16384x1024 @ 1024x3072), bf16 WMMA, bf16 out.
// Wave computes a 16x64 output slab (1 M-tile x 4 N-tiles). A converted
// fp32->bf16 on the fly; B pre-transposed n-major so per-lane loads are
// contiguous 32B.
// ---------------------------------------------------------------------------
__global__ __launch_bounds__(256) void gemm_in_wmma(
    const float* __restrict__ X,     // [16384][1024]
    const __bf16* __restrict__ WT,   // [3072][1024] n-major
    const float* __restrict__ bias,  // [3072]
    __bf16* __restrict__ U) {        // [16384][3072]
  const unsigned lane = threadIdx.x & 31u;
  const unsigned wv = threadIdx.x >> 5;
  const unsigned W = blockIdx.x * 8u + wv;
  const int nt = (int)(W % 48u);
  const int mt = (int)(W / 48u);
  const int m0 = mt * 16, n0 = nt * 64;
  const int MN = (int)(lane & 15u);  // A row / B column / C column
  const bool lo = lane < 16u;
  const int q0 = lo ? 0 : 8;
  const int KH = lo ? 0 : 16;
  v8f a0 = {0, 0, 0, 0, 0, 0, 0, 0}, a1 = a0, a2 = a0, a3 = a0;
  const float* xrow = X + (size_t)(m0 + MN) * Dz + KH;
  const __bf16* brow = WT + (size_t)(n0 + MN) * Dz + KH;
  for (int k0 = 0; k0 < Dz; k0 += 32) {
    __builtin_prefetch(xrow + k0 + 32, 0, 1);
    v16bf a;
#pragma unroll
    for (int e = 0; e < 8; ++e) {
      a[e] = (__bf16)xrow[k0 + e];
      a[e + 8] = (__bf16)xrow[k0 + 8 + e];
    }
    v16bf b0 = *(const v16bf*)(brow + k0);
    v16bf b1 = *(const v16bf*)(brow + k0 + 16 * Dz);
    v16bf b2 = *(const v16bf*)(brow + k0 + 32 * Dz);
    v16bf b3 = *(const v16bf*)(brow + k0 + 48 * Dz);
    a0 = wmma_bf16(a, b0, a0);
    a1 = wmma_bf16(a, b1, a1);
    a2 = wmma_bf16(a, b2, a2);
    a3 = wmma_bf16(a, b3, a3);
  }
  const int nb = n0 + MN;
  const int mhi = lo ? 0 : 8;
  const float bb0 = bias[nb], bb1 = bias[nb + 16], bb2 = bias[nb + 32],
              bb3 = bias[nb + 48];
#pragma unroll
  for (int p = 0; p < 8; ++p) {
    __bf16* urow = U + (size_t)(m0 + p + mhi) * WIDTHz;
    urow[nb] = (__bf16)(a0[p] + bb0);
    urow[nb + 16] = (__bf16)(a1[p] + bb1);
    urow[nb + 32] = (__bf16)(a2[p] + bb2);
    urow[nb + 48] = (__bf16)(a3[p] + bb3);
  }
}

// ---------------------------------------------------------------------------
// K3: depthwise 3-tap causal conv + gate. Produces fp32 x0 (time-major) and
// bf16 v (channel-major, 256-zero front pad) for the Toeplitz WMMA kernel.
// ---------------------------------------------------------------------------
__global__ __launch_bounds__(256) void shortconv_gate(
    const __bf16* __restrict__ U,   // [16384][3072]
    const float* __restrict__ sw,   // [3072][3]
    const float* __restrict__ sb,   // [3072]
    float* __restrict__ x0f,        // [16384][1024]
    __bf16* __restrict__ vp) {      // [4096 ch][4352]
  int idx = blockIdx.x * 256 + threadIdx.x;  // 16.7M threads
  int d = idx & (Dz - 1);
  int t = (idx >> 10) & (Lz - 1);
  int b = idx >> 22;
  int m = b * Lz + t;
  const __bf16* r0 = U + (size_t)m * WIDTHz;
  float res[3];
#pragma unroll
  for (int part = 0; part < 3; ++part) {
    int c = part * Dz + d;
    float u0 = (float)r0[c];
    float u1 = (t >= 1) ? (float)r0[c - WIDTHz] : 0.0f;
    float u2 = (t >= 2) ? (float)r0[c - 2 * WIDTHz] : 0.0f;
    res[part] = u2 * sw[c * 3 + 0] + u1 * sw[c * 3 + 1] + u0 * sw[c * 3 + 2] + sb[c];
  }
  float vv = res[2] * res[1];
  x0f[(size_t)m * Dz + d] = res[0];
  __bf16* vrow = vp + (size_t)(b * Dz + d) * VPz;
  vrow[256 + t] = (__bf16)vv;
  if (t < 256) vrow[t] = (__bf16)0.0f;  // causal/left zero pad
}

// ---------------------------------------------------------------------------
// K4: long causal conv as Toeplitz-blocked WMMA.
// One wave handles channel (b,d) and a group of 16 output tiles i0..i0+15.
//   A (16x32, shared across lanes' columns): A[tt,q] = krev[(4095-32r-tt)+q]
//   B (32x16): column N reads 32 contiguous v values at a per-column shift
//   C (16x16): column N = output tile i0+N, row = t within tile
// Epilogue fuses +filter_bias and the *x0 gate; writes bf16 A for GEMM-out.
// ---------------------------------------------------------------------------
__global__ __launch_bounds__(256) void hyena_toeplitz_wmma(
    const __bf16* __restrict__ krev,  // [1024][4128]
    const __bf16* __restrict__ vp,    // [4096 ch][4352]
    const float* __restrict__ x0f,    // [16384][1024]
    const float* __restrict__ fbias,  // [1024]
    __bf16* __restrict__ Wg) {        // [16384][1024]
  const unsigned lane = threadIdx.x & 31u;
  const unsigned wv = threadIdx.x >> 5;
  const unsigned Wd = blockIdx.x * 8u + wv;
  const int g = (int)(Wd & 15u);
  const int d = (int)((Wd >> 4) & 1023u);
  const int b = (int)(Wd >> 14);
  const int i0 = g * 16;
  const int MN = (int)(lane & 15u);  // A row tt, B/C column N
  const bool lo = lane < 16u;
  const int q0 = lo ? 0 : 8;
  const int KH = lo ? 0 : 16;
  const __bf16* kd = krev + (size_t)d * KPz;
  const __bf16* vd = vp + (size_t)(b * Dz + d) * VPz;
  const int Rmax = 8 * (g + 1);
  // B base (with 256-elem pad): 256 + 16*(i0+N) + 16 + KH - 32*(r+1)
  const int bbase = 272 + 16 * (i0 + MN) + KH;
  v8f acc = {0, 0, 0, 0, 0, 0, 0, 0};
  for (int r = 0; r < Rmax; ++r) {
    int abase = 4095 - 32 * r - MN;
    v8bf h0, h1;
    __builtin_memcpy(&h0, kd + abase + KH, 16);       // K run {0..7}/{16..23}
    __builtin_memcpy(&h1, kd + abase + KH + 8, 16);   // K run {8..15}/{24..31}
    v16bf a = cat8(h0, h1);
    v16bf bf = *(const v16bf*)(vd + bbase - 32 * (r + 1));  // 32B aligned
    acc = wmma_bf16(a, bf, acc);
  }
  const float fb = fbias[d];
  const int mhi = lo ? 0 : 8;
  const int tcol = 16 * (i0 + MN);
#pragma unroll
  for (int p = 0; p < 8; ++p) {
    int t = tcol + p + mhi;
    size_t idx = (size_t)(b * Lz + t) * Dz + d;
    float val = (acc[p] + fb) * x0f[idx];
    Wg[idx] = (__bf16)val;
  }
}

// ---------------------------------------------------------------------------
// K5: y = Wg @ W_out + b_out  (16384x1024 @ 1024x1024), bf16 WMMA, fp32 out.
// ---------------------------------------------------------------------------
__global__ __launch_bounds__(256) void gemm_out_wmma(
    const __bf16* __restrict__ A,    // [16384][1024]
    const __bf16* __restrict__ WT,   // [1024][1024] n-major
    const float* __restrict__ bias,  // [1024]
    float* __restrict__ Y) {         // [16384][1024]
  const unsigned lane = threadIdx.x & 31u;
  const unsigned wv = threadIdx.x >> 5;
  const unsigned W = blockIdx.x * 8u + wv;
  const int ng = (int)(W & 15u);
  const int mt = (int)(W >> 4);
  const int m0 = mt * 16, n0 = ng * 64;
  const int MN = (int)(lane & 15u);
  const bool lo = lane < 16u;
  const int q0 = lo ? 0 : 8;
  const int KH = lo ? 0 : 16;
  v8f a0 = {0, 0, 0, 0, 0, 0, 0, 0}, a1 = a0, a2 = a0, a3 = a0;
  const __bf16* arow = A + (size_t)(m0 + MN) * Dz + KH;
  const __bf16* brow = WT + (size_t)(n0 + MN) * Dz + KH;
  for (int k0 = 0; k0 < Dz; k0 += 32) {
    __builtin_prefetch(arow + k0 + 32, 0, 1);
    v8bf h0 = *(const v8bf*)(arow + k0);
    v8bf h1 = *(const v8bf*)(arow + k0 + 8);
    v16bf a = cat8(h0, h1);
    v16bf b0 = *(const v16bf*)(brow + k0);
    v16bf b1 = *(const v16bf*)(brow + k0 + 16 * Dz);
    v16bf b2 = *(const v16bf*)(brow + k0 + 32 * Dz);
    v16bf b3 = *(const v16bf*)(brow + k0 + 48 * Dz);
    a0 = wmma_bf16(a, b0, a0);
    a1 = wmma_bf16(a, b1, a1);
    a2 = wmma_bf16(a, b2, a2);
    a3 = wmma_bf16(a, b3, a3);
  }
  const int nb = n0 + MN;
  const int mhi = lo ? 0 : 8;
  const float bb0 = bias[nb], bb1 = bias[nb + 16], bb2 = bias[nb + 32],
              bb3 = bias[nb + 48];
#pragma unroll
  for (int p = 0; p < 8; ++p) {
    float* yrow = Y + (size_t)(m0 + p + mhi) * Dz;
    yrow[nb] = a0[p] + bb0;
    yrow[nb + 16] = a1[p] + bb1;
    yrow[nb + 32] = a2[p] + bb2;
    yrow[nb + 48] = a3[p] + bb3;
  }
}

// ---------------------------------------------------------------------------
// Workspace layout (bytes), total ~254 MB
// ---------------------------------------------------------------------------
#define OFF_U     ((size_t)0)           // bf16 u      [16384][3072]  96 MB
#define OFF_WINT  ((size_t)100663296)   // bf16 WinT   [3072][1024]    6 MB
#define OFF_WOUTT ((size_t)106954752)   // bf16 WoutT  [1024][1024]    2 MB
#define OFF_KREV  ((size_t)109051904)   // bf16 krev   [1024][4128]    8 MB
#define OFF_VP    ((size_t)117506048)   // bf16 vp     [4096][4352]   34 MB
#define OFF_X0    ((size_t)153157632)   // f32  x0     [16384][1024]  64 MB
#define OFF_WG    ((size_t)220266496)   // bf16 Wg     [16384][1024]  32 MB

extern "C" void kernel_launch(void* const* d_in, const int* in_sizes, int n_in,
                              void* d_out, int out_size, void* d_ws, size_t ws_size,
                              hipStream_t stream) {
  const float* x        = (const float*)d_in[0];
  const float* W_in     = (const float*)d_in[1];
  const float* b_in     = (const float*)d_in[2];
  const float* short_w  = (const float*)d_in[3];
  const float* short_b  = (const float*)d_in[4];
  const float* W1       = (const float*)d_in[5];
  const float* b1       = (const float*)d_in[6];
  const float* f1       = (const float*)d_in[7];
  const float* W2       = (const float*)d_in[8];
  const float* b2       = (const float*)d_in[9];
  const float* f2       = (const float*)d_in[10];
  const float* W3       = (const float*)d_in[11];
  const float* b3       = (const float*)d_in[12];
  const float* fbias    = (const float*)d_in[13];
  const float* W_out    = (const float*)d_in[14];
  const float* b_out    = (const float*)d_in[15];
  (void)in_sizes; (void)n_in; (void)out_size; (void)ws_size;

  char* ws = (char*)d_ws;
  __bf16* u     = (__bf16*)(ws + OFF_U);
  __bf16* WinT  = (__bf16*)(ws + OFF_WINT);
  __bf16* WoutT = (__bf16*)(ws + OFF_WOUTT);
  __bf16* krev  = (__bf16*)(ws + OFF_KREV);
  __bf16* vp    = (__bf16*)(ws + OFF_VP);
  float*  x0f   = (float*)(ws + OFF_X0);
  __bf16* wg    = (__bf16*)(ws + OFF_WG);
  float* out = (float*)d_out;

  transpose_cvt<<<(Dz * WIDTHz) / 256, 256, 0, stream>>>(W_in, WinT, Dz, WIDTHz);
  transpose_cvt<<<(Dz * Dz) / 256, 256, 0, stream>>>(W_out, WoutT, Dz, Dz);
  gen_filter<<<Lz, 256, 0, stream>>>(W1, b1, f1, W2, b2, f2, W3, b3, krev);
  gemm_in_wmma<<<6144, 256, 0, stream>>>(x, WinT, b_in, u);          // 49152 waves
  shortconv_gate<<<(Bz * Lz * Dz) / 256, 256, 0, stream>>>(u, short_w, short_b,
                                                           x0f, vp);
  hyena_toeplitz_wmma<<<8192, 256, 0, stream>>>(krev, vp, x0f, fbias, wg);
  gemm_out_wmma<<<2048, 256, 0, stream>>>(wg, WoutT, b_out, out);    // 16384 waves
}
